// MoDE_22643067585077
// MI455X (gfx1250) — compile-verified, matrix operand order checked by reading, not run on hardware
//
#include <hip/hip_runtime.h>
#include <hip/hip_bf16.h>

typedef __bf16 bf16;
typedef __attribute__((ext_vector_type(16))) __bf16 v16bf;
typedef __attribute__((ext_vector_type(8)))  float  v8f;

// Builtin-compatible pointee type for global_load_async_to_lds_b128:
// param is 'int __vector(4) addrspace(1)*' / 'addrspace(3)*'
typedef int v4i_gcc __attribute__((vector_size(16)));
typedef __attribute__((address_space(1))) v4i_gcc as1_v4i;
typedef __attribute__((address_space(3))) v4i_gcc as3_v4i;

#define B_   32
#define CI_  64
#define CO_  64
#define H_   128
#define W_   128
#define NT_  10
#define NE_  5

// Workspace layout (bytes):
//   Wbf : [B][25][CO][CI] bf16   -> 32*25*64*64*2 = 6,553,600
//   Xt  : [B][H][W][CI]   bf16   -> 32*128*128*64*2 = 67,108,864
#define WS_WBF_BYTES 6553600ull

// LDS slab geometry for the GEMM kernel: [5 rows][132 padded cols][32 ci] bf16
#define SLAB_ELEMS   (5 * 132 * 32)      // 21120 bf16 = 42,240 B
#define SLAB_SLOTS   2640                // SLAB_ELEMS / 8 (uint4 slots)
#define DUMP_OFF     (2 * SLAB_ELEMS)    // 16B scratch slot for masked lanes

#if defined(__has_builtin)
#if __has_builtin(__builtin_amdgcn_global_load_async_to_lds_b128) && \
    __has_builtin(__builtin_amdgcn_s_wait_asynccnt)
#define USE_ASYNC_LDS 1
#endif
#endif

// ---------------------------------------------------------------------------
// Kernel 0: x NCHW fp32 -> NHWC bf16 (ci innermost) via LDS tile transpose.
// One block per (b, h); coalesced reads and writes.
// ---------------------------------------------------------------------------
__global__ __launch_bounds__(256) void mode_nchw_to_nhwc_bf16(
    const float* __restrict__ x, bf16* __restrict__ Xt)
{
    __shared__ __align__(16) bf16 tile[CI_ * 132];  // stride 132 to spread banks
    const int h = blockIdx.x;
    const int b = blockIdx.y;
    for (int i = threadIdx.x; i < CI_ * W_; i += 256) {
        int ci = i >> 7, w = i & 127;                 // consecutive tid -> consecutive w
        tile[ci * 132 + w] =
            (bf16)x[(((size_t)b * CI_ + ci) * H_ + h) * W_ + w];
    }
    __syncthreads();
    for (int i = threadIdx.x; i < W_ * CI_; i += 256) {
        int w = i >> 6, ci = i & 63;                  // consecutive tid -> consecutive ci
        Xt[(((size_t)b * H_ + h) * W_ + w) * CI_ + ci] = tile[ci * 132 + w];
    }
}

// ---------------------------------------------------------------------------
// Kernel 1: gate softmax + per-sample kernel synthesis into Wbf[b][tap][o][ci]
// (tap-major, ci innermost => contiguous WMMA A-fragments). Also emits task_id.
// ---------------------------------------------------------------------------
__global__ __launch_bounds__(256) void mode_gate_weights(
    const int*   __restrict__ task_id,
    const float* __restrict__ gate_w,  // [NE*CO][NT]
    const float* __restrict__ gate_b,  // [NE*CO]
    const float* __restrict__ w5,      // [CO][CI][5][5]
    const float* __restrict__ w3,      // [CO][CI][3][3]
    const float* __restrict__ w1,      // [CO][CI]
    const float* __restrict__ wavg3,   // [CO][CI]
    const float* __restrict__ wavg5,   // [CO][CI]
    bf16* __restrict__ Wbf,
    int*  __restrict__ task_out)
{
    const int b = blockIdx.x;
    const int t = task_id[b];
    if (threadIdx.x == 0) task_out[b] = t;   // second tuple output (identity)

    for (int p = threadIdx.x; p < CO_ * CI_; p += 256) {
        const int o  = p >> 6;
        const int ci = p & 63;

        // softmax over 5 experts for (b, o)
        float g[NE_];
        float m = -1e30f;
        #pragma unroll
        for (int e = 0; e < NE_; ++e) {
            float l = gate_w[(e * CO_ + o) * NT_ + t] + gate_b[e * CO_ + o];
            g[e] = l;
            m = fmaxf(m, l);
        }
        float s = 0.f;
        #pragma unroll
        for (int e = 0; e < NE_; ++e) { g[e] = __expf(g[e] - m); s += g[e]; }
        const float inv = 1.f / s;
        #pragma unroll
        for (int e = 0; e < NE_; ++e) g[e] *= inv;

        const float* p5  = w5 + (size_t)p * 25;
        const float* p3  = w3 + (size_t)p * 9;
        const float  v1  = w1[p];
        const float  va3 = wavg3[p] * (1.f / 9.f);
        const float  va5 = wavg5[p] * (1.f / 25.f);

        #pragma unroll
        for (int ky = 0; ky < 5; ++ky) {
            #pragma unroll
            for (int kx = 0; kx < 5; ++kx) {
                float val = g[0] * p5[ky * 5 + kx] + g[4] * va5;
                if (ky >= 1 && ky <= 3 && kx >= 1 && kx <= 3)
                    val += g[1] * p3[(ky - 1) * 3 + (kx - 1)] + g[3] * va3;
                if (ky == 2 && kx == 2)
                    val += g[2] * v1;
                const int tap = ky * 5 + kx;
                Wbf[((((size_t)b * 25 + tap) * CO_) + o) * CI_ + ci] = (bf16)val;
            }
        }
    }
}

// ---------------------------------------------------------------------------
// Kernel 2: implicit-GEMM conv row. One block = (b, output row h).
// GEMM: M=64 oc, N=128 width, K=25 taps * 64 ci, via v_wmma_f32_16x16x32_bf16.
// 8 waves: 4 along M x 2 along N; each wave owns 4 fp32 16x16 accumulators.
// Double-buffered LDS slabs (2 x [5][132][32ci] bf16 = 84,480 B) filled with
// GLOBAL_LOAD_ASYNC_TO_LDS_B128 (ASYNCcnt) so chunk-1 staging overlaps
// chunk-0 WMMA compute. Falls back to synchronous staging if the async
// builtins are unavailable on this toolchain.
// ---------------------------------------------------------------------------
__global__ __launch_bounds__(256) void mode_conv_wmma(
    const bf16* __restrict__ Xt,    // [B][H][W][CI]
    const bf16* __restrict__ Wbf,   // [B][25][CO][CI]
    float* __restrict__ y)          // [B][CO][H][W]
{
    __shared__ __align__(16) bf16 slab[2 * SLAB_ELEMS + 8];  // +16B dump slot

    const int h    = blockIdx.x;
    const int b    = blockIdx.y;
    const int tid  = threadIdx.x;
    const int lane = tid & 31;          // wave32
    const int wave = tid >> 5;
    const int half = lane >> 4;         // K-group selector (ISA 16-bit A layout)
    const int idx  = lane & 15;         // M (A) / N (B,C) index within tile
    const int m_base = (wave & 3) << 4; // oc tile base: 0,16,32,48
    const int n_base = (wave >> 2) << 6;// width tile base: 0 or 64

    // ---- step 1: pre-zero halo / out-of-image slots of BOTH buffers -------
    for (int u = tid; u < 2 * SLAB_SLOTS; u += 256) {
        const int buf = (u >= SLAB_SLOTS) ? 1 : 0;
        const int t   = u - buf * SLAB_SLOTS;
        const int q   = t & 3;
        const int rc  = t >> 2;
        const int c   = rc % 132;
        const int r   = rc / 132;
        const int win = c - 2;
        const int hin = h + r - 2;
        if (!((unsigned)win < 128u && (unsigned)hin < 128u)) {
            *(uint4*)(slab + (size_t)buf * SLAB_ELEMS + ((size_t)r * 132 + c) * 32 + q * 8) =
                make_uint4(0u, 0u, 0u, 0u);
        }
    }

    // ---- step 2: fill valid slots of both buffers --------------------------
#if defined(USE_ASYNC_LDS)
    // Uniform issue: every thread fires exactly 11 async b128 loads per buffer
    // (masked lanes target the dump slot with an in-range clamped source), so
    // the per-wave ASYNCcnt is deterministic: wait <=11 => buffer 0 complete.
    #pragma unroll
    for (int buf = 0; buf < 2; ++buf) {
        for (int k = 0; k < 11; ++k) {
            const int u   = tid + k * 256;
            const int t   = (u < SLAB_SLOTS) ? u : 0;
            const int q   = t & 3;
            const int rc  = t >> 2;
            const int c   = rc % 132;
            const int r   = rc / 132;
            const int win = c - 2;
            const int hin = h + r - 2;
            const bool ok = (u < SLAB_SLOTS) &&
                            ((unsigned)win < 128u) && ((unsigned)hin < 128u);
            const int hc  = min(max(hin, 0), 127);
            const int wc  = min(max(win, 0), 127);
            const bf16* src = Xt + (((size_t)b * H_ + hc) * W_ + wc) * CI_ +
                              buf * 32 + q * 8;
            const bf16* dst = slab + (ok
                ? ((size_t)buf * SLAB_ELEMS + ((size_t)r * 132 + c) * 32 + q * 8)
                : (size_t)DUMP_OFF);
            __builtin_amdgcn_global_load_async_to_lds_b128(
                (as1_v4i*)(unsigned long long)src,
                (as3_v4i*)(unsigned)(unsigned long long)dst,
                0, 0);
        }
    }
    __builtin_amdgcn_s_wait_asynccnt(11);   // buffer 0 landed; buffer 1 in flight
#else
    for (int u = tid; u < 2 * SLAB_SLOTS; u += 256) {
        const int buf = (u >= SLAB_SLOTS) ? 1 : 0;
        const int t   = u - buf * SLAB_SLOTS;
        const int q   = t & 3;
        const int rc  = t >> 2;
        const int c   = rc % 132;
        const int r   = rc / 132;
        const int win = c - 2;
        const int hin = h + r - 2;
        if ((unsigned)win < 128u && (unsigned)hin < 128u) {
            *(uint4*)(slab + (size_t)buf * SLAB_ELEMS + ((size_t)r * 132 + c) * 32 + q * 8) =
                *(const uint4*)(Xt + (((size_t)b * H_ + hin) * W_ + win) * CI_ +
                                buf * 32 + q * 8);
        }
    }
#endif
    __syncthreads();

    v8f acc0 = {}, acc1 = {}, acc2 = {}, acc3 = {};

    for (int cc = 0; cc < 2; ++cc) {        // two 32-ci K-chunks
        const bf16* sb = slab + (size_t)cc * SLAB_ELEMS;

        for (int tap = 0; tap < 25; ++tap) {
            const int ky = tap / 5;
            const int kx = tap % 5;

            // A fragment: Wbf[b][tap][m_base+idx][cc*32 + half*8 .. +7, +16..]
            const bf16* aptr = Wbf +
                ((((size_t)b * 25 + tap) * CO_) + (m_base + idx)) * CI_ +
                cc * 32 + half * 8;
            union { uint4 u[2]; v16bf v; } af;
            af.u[0] = *(const uint4*)(aptr);
            af.u[1] = *(const uint4*)(aptr + 16);
            if (tap + 1 < 25)
                __builtin_prefetch((const void*)(aptr + (size_t)CO_ * CI_), 0, 0);

            #pragma unroll
            for (int nt = 0; nt < 4; ++nt) {
                const int n = n_base + nt * 16 + idx;       // output column
                const bf16* bptr = sb +
                    ((size_t)(ky * 132 + (n + kx))) * 32 + half * 8;
                union { uint4 u[2]; v16bf v; } bfm;
                bfm.u[0] = *(const uint4*)(bptr);           // ci = half*8..+7
                bfm.u[1] = *(const uint4*)(bptr + 16);      // ci = 16+half*8..+7
                v8f* accp = (nt == 0) ? &acc0 : (nt == 1) ? &acc1
                          : (nt == 2) ? &acc2 : &acc3;
                *accp = __builtin_amdgcn_wmma_f32_16x16x32_bf16(
                    false, af.v, false, bfm.v, (short)0, *accp, false, false);
            }
        }

        if (cc == 0) {
#if defined(USE_ASYNC_LDS)
            __builtin_amdgcn_s_wait_asynccnt(0);  // buffer 1 fully landed
#endif
            __syncthreads();
        }
    }

    // ---- store C: vgpr r -> oc = m_base + half*8 + r; N = lane&15 ----------
    #pragma unroll
    for (int nt = 0; nt < 4; ++nt) {
        const v8f a = (nt == 0) ? acc0 : (nt == 1) ? acc1
                    : (nt == 2) ? acc2 : acc3;
        const int wout = n_base + nt * 16 + idx;
        #pragma unroll
        for (int r = 0; r < 8; ++r) {
            const int oc = m_base + half * 8 + r;
            y[(((size_t)b * CO_ + oc) * H_ + h) * W_ + wout] = a[r];
        }
    }
}

// ---------------------------------------------------------------------------
extern "C" void kernel_launch(void* const* d_in, const int* in_sizes, int n_in,
                              void* d_out, int out_size, void* d_ws, size_t ws_size,
                              hipStream_t stream)
{
    const float* x       = (const float*)d_in[0];
    const int*   task_id = (const int*)  d_in[1];
    const float* gate_w  = (const float*)d_in[2];
    const float* gate_b  = (const float*)d_in[3];
    const float* w5      = (const float*)d_in[4];
    const float* w3      = (const float*)d_in[5];
    const float* w1      = (const float*)d_in[6];
    const float* wavg3   = (const float*)d_in[7];
    const float* wavg5   = (const float*)d_in[8];

    float* y = (float*)d_out;
    int* task_out = (int*)(y + (size_t)B_ * CO_ * H_ * W_);  // tuple tail

    bf16* Wbf = (bf16*)d_ws;
    bf16* Xt  = (bf16*)((char*)d_ws + WS_WBF_BYTES);

    dim3 blk(256);
    mode_nchw_to_nhwc_bf16<<<dim3(H_, B_), blk, 0, stream>>>(x, Xt);
    mode_gate_weights<<<dim3(B_), blk, 0, stream>>>(
        task_id, gate_w, gate_b, w5, w3, w1, wavg3, wavg5, Wbf, task_out);
    mode_conv_wmma<<<dim3(H_, B_), blk, 0, stream>>>(Xt, Wbf, y);
}
